// CLIPSum_26001732010423
// MI455X (gfx1250) — compile-verified, matrix operand order checked by reading, not run on hardware
//
#include <hip/hip_runtime.h>
#include <hip/hip_bf16.h>

namespace {
constexpr int B_    = 64;
constexpr int VLEN  = 256;
constexpr int TLEN  = 512;
constexpr int HDIM  = 512;
constexpr int KVv   = (VLEN - 1) / 2;   // 127
constexpr int KTt   = (TLEN - 1) / 2;   // 255
constexpr int HALF_CNT = (VLEN * TLEN) / 2; // 65536
constexpr int NBINS = 2048;
constexpr float EPSf = 1e-8f;
constexpr int NTHREADS = 256;
}

typedef __attribute__((ext_vector_type(16))) __bf16 bf16x16;
typedef __attribute__((ext_vector_type(8)))  __bf16 bf16x8;
typedef __attribute__((ext_vector_type(8)))  float  f32x8;

// Order-preserving float -> uint map so atomicMax works for signed floats.
__device__ __forceinline__ unsigned ordmap(float f) {
  unsigned u = __float_as_uint(f);
  return (u & 0x80000000u) ? ~u : (u | 0x80000000u);
}

__global__ void zero_u32_kernel(unsigned* __restrict__ p, int n) {
  int i = blockIdx.x * blockDim.x + threadIdx.x;
  if (i < n) p[i] = 0u;
}

// One wave (32 lanes) per embedding row: ||row||_2.
__global__ void norms_kernel(const float* __restrict__ V, const float* __restrict__ T,
                             float* __restrict__ vn, float* __restrict__ tn) {
  int row  = blockIdx.x * 8 + (threadIdx.x >> 5);
  int lane = threadIdx.x & 31;
  const int NV = B_ * VLEN;
  const float* p = (row < NV) ? (V + (size_t)row * HDIM)
                              : (T + (size_t)(row - NV) * HDIM);
  float s = 0.f;
  for (int i = lane * 4; i < HDIM; i += 128) {
    float4 x = *reinterpret_cast<const float4*>(p + i);
    s += x.x * x.x + x.y * x.y + x.z * x.z + x.w * x.w;
  }
#pragma unroll
  for (int d = 16; d > 0; d >>= 1) s += __shfl_xor(s, d, 32);
  if (lane == 0) {
    float nv = sqrtf(s);
    if (row < NV) vn[row] = nv; else tn[row - NV] = nv;
  }
}

// Load a 16x32 bf16 A/B fragment from an LDS tile row per the CDNA5 WMMA
// VGPR layout: lanes 0-15 hold K={0..7,16..23}, lanes 16-31 hold K={8..15,24..31}.
__device__ __forceinline__ bf16x16 load_frag(const __bf16* rowBase, int kHalf) {
  union { bf16x16 v; bf16x8 h[2]; } u;
  u.h[0] = *reinterpret_cast<const bf16x8*>(rowBase + kHalf);
  u.h[1] = *reinterpret_cast<const bf16x8*>(rowBase + kHalf + 16);
  return u.v;
}

// 128x128 C-tile per workgroup; 8 waves in 2(M)x4(N); each wave owns 64x32 = 4x2
// WMMA accumulators. Split-bf16 fp32 emulation: hi*hi + hi*lo + lo*hi.
__launch_bounds__(NTHREADS, 1)
__global__ void simstat_kernel(const float* __restrict__ V, const float* __restrict__ T,
                               const float* __restrict__ vn, const float* __restrict__ tn,
                               unsigned* __restrict__ rowmax, unsigned* __restrict__ colmax,
                               unsigned* __restrict__ hist) {
  const int b = blockIdx.z, mt = blockIdx.y, nt = blockIdx.x;
  const int tid  = threadIdx.x;
  const int lane = tid & 31, w = tid >> 5;
  const int wm = w >> 2, wn = w & 3;

  __shared__ __bf16 Ahi[128][32], Alo[128][32], Bhi[128][32], Blo[128][32];
  __shared__ float snv[128], snt[128];
  __shared__ unsigned sRow[128], sCol[128], sHist[NBINS];

  const float* vB = V + ((size_t)b * VLEN + mt * 128) * HDIM;
  const float* tB = T + ((size_t)b * TLEN + nt * 128) * HDIM;

  if (tid < 128) {
    snv[tid] = vn[b * VLEN + mt * 128 + tid];
    snt[tid] = tn[b * TLEN + nt * 128 + tid];
    sRow[tid] = 0u; sCol[tid] = 0u;
  }
  for (int i = tid; i < NBINS; i += NTHREADS) sHist[i] = 0u;

  const f32x8 vzero = {0.f, 0.f, 0.f, 0.f, 0.f, 0.f, 0.f, 0.f};
  f32x8 acc[4][2];
#pragma unroll
  for (int mi = 0; mi < 4; ++mi)
#pragma unroll
    for (int ni = 0; ni < 2; ++ni) acc[mi][ni] = vzero;

  for (int k0 = 0; k0 < HDIM; k0 += 32) {
    __syncthreads();
    // Stage 128x32 fp32 slabs of A and B as bf16 hi + bf16 residual in LDS.
#pragma unroll
    for (int i = 0; i < 4; ++i) {
      int idx = tid + i * NTHREADS;   // 0..1023 float4 slots
      int r   = idx >> 3;
      int c   = (idx & 7) << 2;
      float4 a  = *reinterpret_cast<const float4*>(vB + (size_t)r * HDIM + k0 + c);
      float4 t4 = *reinterpret_cast<const float4*>(tB + (size_t)r * HDIM + k0 + c);
      float av[4] = {a.x, a.y, a.z, a.w};
      float tv[4] = {t4.x, t4.y, t4.z, t4.w};
#pragma unroll
      for (int q = 0; q < 4; ++q) {
        __bf16 ah = (__bf16)av[q];
        Ahi[r][c + q] = ah;
        Alo[r][c + q] = (__bf16)(av[q] - (float)ah);
        __bf16 th = (__bf16)tv[q];
        Bhi[r][c + q] = th;
        Blo[r][c + q] = (__bf16)(tv[q] - (float)th);
      }
    }
    __syncthreads();

    const int kHalf = (lane < 16) ? 0 : 8;
    bf16x16 aHi[4], aLo[4], bHi[2], bLo[2];
#pragma unroll
    for (int mi = 0; mi < 4; ++mi) {
      int r = wm * 64 + mi * 16 + (lane & 15);
      aHi[mi] = load_frag(&Ahi[r][0], kHalf);
      aLo[mi] = load_frag(&Alo[r][0], kHalf);
    }
#pragma unroll
    for (int ni = 0; ni < 2; ++ni) {
      int r = wn * 32 + ni * 16 + (lane & 15);
      bHi[ni] = load_frag(&Bhi[r][0], kHalf);
      bLo[ni] = load_frag(&Blo[r][0], kHalf);
    }
#pragma unroll
    for (int mi = 0; mi < 4; ++mi)
#pragma unroll
      for (int ni = 0; ni < 2; ++ni) {
        acc[mi][ni] = __builtin_amdgcn_wmma_f32_16x16x32_bf16(
            false, aHi[mi], false, bHi[ni], (short)0, acc[mi][ni], false, false);
        acc[mi][ni] = __builtin_amdgcn_wmma_f32_16x16x32_bf16(
            false, aHi[mi], false, bLo[ni], (short)0, acc[mi][ni], false, false);
        acc[mi][ni] = __builtin_amdgcn_wmma_f32_16x16x32_bf16(
            false, aLo[mi], false, bHi[ni], (short)0, acc[mi][ni], false, false);
      }
  }
  __syncthreads();

  // Epilogue: cosine divide, then LDS-aggregated row/col maxes + histogram.
#pragma unroll
  for (int mi = 0; mi < 4; ++mi) {
#pragma unroll
    for (int ni = 0; ni < 2; ++ni) {
      int nl = wn * 32 + ni * 16 + (lane & 15);
      float tnv = snt[nl];
#pragma unroll
      for (int j = 0; j < 8; ++j) {
        int ml = wm * 64 + mi * 16 + j + ((lane >> 4) << 3);
        float sim = acc[mi][ni][j] / fmaxf(snv[ml] * tnv, EPSf);
        unsigned u = ordmap(sim);
        atomicMax(&sRow[ml], u);
        atomicMax(&sCol[nl], u);
        int bin = (int)((sim + 1.0f) * (NBINS * 0.5f));
        bin = bin < 0 ? 0 : (bin > NBINS - 1 ? NBINS - 1 : bin);
        atomicAdd(&sHist[bin], 1u);
      }
    }
  }
  __syncthreads();
  if (tid < 128) {
    atomicMax(&rowmax[b * VLEN + mt * 128 + tid], sRow[tid]);
    atomicMax(&colmax[b * TLEN + nt * 128 + tid], sCol[tid]);
  }
  for (int i = tid; i < NBINS; i += NTHREADS)
    if (sHist[i]) atomicAdd(&hist[b * NBINS + i], sHist[i]);
}

// Per-batch: median threshold from histogram, rank rows/cols by descending max,
// build the _split_rows destination maps (mutual padded with -1).
__launch_bounds__(NTHREADS, 1)
__global__ void select_kernel(const unsigned* __restrict__ rowmax,
                              const unsigned* __restrict__ colmax,
                              const unsigned* __restrict__ hist,
                              int* __restrict__ mVm, int* __restrict__ mVo,
                              int* __restrict__ mTm, int* __restrict__ mTo) {
  int b = blockIdx.x, tid = threadIdx.x;
  __shared__ unsigned hRow[VLEN];
  __shared__ unsigned hCol[TLEN];
  __shared__ unsigned char selV[VLEN];
  __shared__ unsigned char selT[TLEN];
  __shared__ unsigned tauU;
  __shared__ int cntV, cntT;

  if (tid < VLEN) hRow[tid] = rowmax[b * VLEN + tid];
  hCol[tid]       = colmax[b * TLEN + tid];
  hCol[tid + 256] = colmax[b * TLEN + tid + 256];
  if (tid == 0) {
    const unsigned* h = hist + b * NBINS;
    long long s = 0; int Tb = 0;
    for (int i = NBINS - 1; i >= 0; --i) { s += h[i]; if (s >= HALF_CNT) { Tb = i; break; } }
    float tau = -1.0f + (2.0f / NBINS) * (float)Tb;  // lower edge of split bin
    tauU = ordmap(tau);
  }
  __syncthreads();

  if (tid < VLEN) {
    unsigned m = hRow[tid]; int rank = 0;
    for (int r = 0; r < VLEN; ++r) { unsigned o = hRow[r]; rank += (o > m) || (o == m && r < tid); }
    selV[tid] = (rank < KVv && m >= tauU) ? 1 : 0;
  }
  for (int c = tid; c < TLEN; c += NTHREADS) {
    unsigned m = hCol[c]; int rank = 0;
    for (int r = 0; r < TLEN; ++r) { unsigned o = hCol[r]; rank += (o > m) || (o == m && r < c); }
    selT[c] = (rank < KTt && m >= tauU) ? 1 : 0;
  }
  __syncthreads();
  if (tid == 0) {
    int s = 0; for (int i = 0; i < VLEN; ++i) s += selV[i]; cntV = s;
    s = 0;     for (int i = 0; i < TLEN; ++i) s += selT[i]; cntT = s;
  }
  __syncthreads();
  const int cV = cntV, cT = cntT;

  if (tid < VLEN) {
    int pre = 0;
    for (int r = 0; r < tid; ++r) pre += selV[r];
    int preU = tid - pre;
    if (selV[tid]) {
      mVm[b * KVv + pre] = tid;
    } else {
      int slot = preU - (KVv - cV);   // reference drops first (k-count) unselected
      if (slot >= 0 && slot < VLEN - KVv) mVo[b * (VLEN - KVv) + slot] = tid;
    }
  }
  if (tid >= cV && tid < KVv) mVm[b * KVv + tid] = -1;  // zero-padding slots

  for (int c = tid; c < TLEN; c += NTHREADS) {
    int pre = 0;
    for (int r = 0; r < c; ++r) pre += selT[r];
    int preU = c - pre;
    if (selT[c]) {
      mTm[b * KTt + pre] = c;
    } else {
      int slot = preU - (KTt - cT);
      if (slot >= 0 && slot < TLEN - KTt) mTo[b * (TLEN - KTt) + slot] = c;
    }
  }
  for (int s = tid; s < KTt; s += NTHREADS) if (s >= cT) mTm[b * KTt + s] = -1;
}

// One block per output row across the 4 concatenated regions
// (v_mut | t_mut | v_only | t_only); 128 threads x float4 = 512 floats.
__global__ void gather_kernel(const float* __restrict__ V, const float* __restrict__ T,
                              const int* __restrict__ mVm, const int* __restrict__ mVo,
                              const int* __restrict__ mTm, const int* __restrict__ mTo,
                              float* __restrict__ out) {
  const int R1 = B_ * KVv, R2 = B_ * KTt, R3 = B_ * (VLEN - KVv);
  int row = blockIdx.x;
  int tid = threadIdx.x;
  const float* srcBase;
  int r;
  if (row < R1) {
    r = mVm[row];
    srcBase = V + (size_t)(row / KVv) * VLEN * HDIM;
  } else if (row < R1 + R2) {
    int q = row - R1; r = mTm[q];
    srcBase = T + (size_t)(q / KTt) * TLEN * HDIM;
  } else if (row < R1 + R2 + R3) {
    int q = row - R1 - R2; r = mVo[q];
    srcBase = V + (size_t)(q / (VLEN - KVv)) * VLEN * HDIM;
  } else {
    int q = row - R1 - R2 - R3; r = mTo[q];
    srcBase = T + (size_t)(q / (TLEN - KTt)) * TLEN * HDIM;
  }
  float4 val = make_float4(0.f, 0.f, 0.f, 0.f);
  if (r >= 0) val = *reinterpret_cast<const float4*>(srcBase + (size_t)r * HDIM + tid * 4);
  *reinterpret_cast<float4*>(out + (size_t)row * HDIM + tid * 4) = val;
}

extern "C" void kernel_launch(void* const* d_in, const int* in_sizes, int n_in,
                              void* d_out, int out_size, void* d_ws, size_t ws_size,
                              hipStream_t stream) {
  (void)in_sizes; (void)n_in; (void)out_size; (void)ws_size;
  const float* V = (const float*)d_in[0];   // [B,256,512]
  const float* T = (const float*)d_in[1];   // [B,512,512]
  float* out = (float*)d_out;

  // Workspace layout (contiguous; rowmax/colmax/hist adjacent for one zero pass)
  char* w = (char*)d_ws;
  float*    vn     = (float*)w;    w += (size_t)B_ * VLEN * 4;
  float*    tn     = (float*)w;    w += (size_t)B_ * TLEN * 4;
  unsigned* rowmax = (unsigned*)w; w += (size_t)B_ * VLEN * 4;
  unsigned* colmax = (unsigned*)w; w += (size_t)B_ * TLEN * 4;
  unsigned* hist   = (unsigned*)w; w += (size_t)B_ * NBINS * 4;
  int* mVm = (int*)w;              w += (size_t)B_ * KVv * 4;
  int* mVo = (int*)w;              w += (size_t)B_ * (VLEN - KVv) * 4;
  int* mTm = (int*)w;              w += (size_t)B_ * KTt * 4;
  int* mTo = (int*)w;              w += (size_t)B_ * (TLEN - KTt) * 4;

  const int nz = B_ * (VLEN + TLEN + NBINS);
  zero_u32_kernel<<<(nz + NTHREADS - 1) / NTHREADS, NTHREADS, 0, stream>>>(rowmax, nz);

  const int nrows = B_ * VLEN + B_ * TLEN;
  norms_kernel<<<nrows / 8, NTHREADS, 0, stream>>>(V, T, vn, tn);

  dim3 g(TLEN / 128, VLEN / 128, B_);
  simstat_kernel<<<g, NTHREADS, 0, stream>>>(V, T, vn, tn, rowmax, colmax, hist);

  select_kernel<<<B_, NTHREADS, 0, stream>>>(rowmax, colmax, hist, mVm, mVo, mTm, mTo);

  const int Rtot = B_ * KVv + B_ * KTt + B_ * (VLEN - KVv) + B_ * (TLEN - KTt);
  gather_kernel<<<Rtot, 128, 0, stream>>>(V, T, mVm, mVo, mTm, mTo, out);
}
